// DifferenceAggregator_62345745269143
// MI455X (gfx1250) — matrix operand order repaired and checked by reading, not compiled, not run on hardware
//
#include <hip/hip_runtime.h>
#include <math.h>

// ---------------- types ----------------
typedef __attribute__((ext_vector_type(16))) __bf16 v16bf;
typedef __attribute__((ext_vector_type(8)))  float  v8f;

#define C_CH   256
#define NPIX   4096
#define BATCH  16
#define S_ELEMS  (BATCH * C_CH * NPIX)        // 16777216
#define SH_ELEMS (BATCH * C_CH * 64 * 58)     // 15204352

__device__ __forceinline__ float sigm_(float x) { return 1.0f / (1.0f + expf(-x)); }
__device__ __forceinline__ float gelu_(float x) { return 0.5f * x * (1.0f + erff(x * 0.70710678118654752440f)); }

// ======================= WMMA GEMM: out(b,M,N) = W(M,K) @ in(b,K,N) + epilogue =======================
#define BM  128
#define BN  128
#define BKK 32
#define LDA 34
#define LDB 34

enum { EPI_BIAS = 0, EPI_BNSILU = 1, EPI_BNSILU_SIG = 2, EPI_BNSILU_SIG_MUL = 3, EPI_GELU = 4 };

template<int EPI, bool SPLIT, bool INBF16, bool OUTBF16>
__global__ __launch_bounds__(256)
void gemm1x1_kernel(const float* __restrict__ Wt,
                    const void* __restrict__ In0v,
                    const void* __restrict__ In1v,
                    void* __restrict__ Outv,
                    const float* __restrict__ p0,
                    const float* __restrict__ p1,
                    int M, int K)
{
    __shared__ __bf16 As[BM * LDA];
    __shared__ __bf16 Bs[BN * LDB];

    const int t    = threadIdx.x;
    const int lane = t & 31;
    const int wave = t >> 5;
    const int bz   = blockIdx.z;
    const int m0   = blockIdx.y * BM;
    const int n0   = blockIdx.x * BN;
    const int wm0  = (wave & 1) * 64;   // 2 m-subtiles of 64
    const int wn0  = (wave >> 1) * 32;  // 4 n-subtiles of 32

    v8f acc[4][2];
    const v8f vzero = {0.f,0.f,0.f,0.f,0.f,0.f,0.f,0.f};
    #pragma unroll
    for (int i = 0; i < 4; ++i) { acc[i][0] = vzero; acc[i][1] = vzero; }

    for (int k0 = 0; k0 < K; k0 += BKK) {
        // ---- stage A (BM x 32) weights fp32 -> bf16 ----
        {
            int m  = t >> 1;
            int kh = (t & 1) * 16;
            const float* src = Wt + (size_t)(m0 + m) * K + k0 + kh;
            __bf16* dst = As + m * LDA + kh;
            #pragma unroll
            for (int i = 0; i < 16; ++i) dst[i] = (__bf16)src[i];
        }
        // ---- stage B (32 x BN) input -> bf16, stored n-major [BN][LDB] ----
        {
            int kk = t >> 3;
            int nc = (t & 7) * 16;
            int c  = k0 + kk;
            if (INBF16) {
                const __bf16* src = (const __bf16*)In0v + ((size_t)bz * K + c) * NPIX + n0 + nc;
                #pragma unroll
                for (int i = 0; i < 16; ++i) Bs[(nc + i) * LDB + kk] = src[i];
            } else {
                const float* src;
                if (SPLIT) {
                    src = (c < 256)
                        ? ((const float*)In0v + ((size_t)bz * 256 + c)       * NPIX + n0 + nc)
                        : ((const float*)In1v + ((size_t)bz * 256 + (c-256)) * NPIX + n0 + nc);
                } else {
                    src = (const float*)In0v + ((size_t)bz * K + c) * NPIX + n0 + nc;
                }
                #pragma unroll
                for (int i = 0; i < 16; ++i) Bs[(nc + i) * LDB + kk] = (__bf16)src[i];
            }
        }
        __syncthreads();

        const int fm = lane & 15;
        const int fh = lane >> 4;
        // B fragments: lane holds col n, contiguous k = fh*16 + e
        v16bf bf0, bf1;
        {
            const __bf16* r0 = &Bs[(wn0 +      fm) * LDB + fh * 16];
            const __bf16* r1 = &Bs[(wn0 + 16 + fm) * LDB + fh * 16];
            #pragma unroll
            for (int e = 0; e < 16; ++e) { bf0[e] = r0[e]; bf1[e] = r1[e]; }
        }
        #pragma unroll
        for (int i = 0; i < 4; ++i) {
            const __bf16* arow = &As[(wm0 + i * 16 + fm) * LDA];
            v16bf af;
            #pragma unroll
            for (int e = 0; e < 16; ++e) {
                int k = (fh << 4) + e;   // half*16 + e
                af[e] = arow[k];
            }
            acc[i][0] = __builtin_amdgcn_wmma_f32_16x16x32_bf16(false, af, false, bf0,
                                                                (short)0, acc[i][0], false, false);
            acc[i][1] = __builtin_amdgcn_wmma_f32_16x16x32_bf16(false, af, false, bf1,
                                                                (short)0, acc[i][1], false, false);
        }
        __syncthreads();
    }

    // ---- epilogue + store ----
    const int fm = lane & 15;
    const int fh = lane >> 4;
    const int nl = wn0 + fm;
    const int mb = wm0 + (fh << 3);
    #pragma unroll
    for (int i = 0; i < 4; ++i) {
        #pragma unroll
        for (int j = 0; j < 2; ++j) {
            int n = n0 + nl + j * 16;
            #pragma unroll
            for (int r = 0; r < 8; ++r) {
                int m = m0 + mb + i * 16 + r;
                float v = acc[i][j][r];
                if (EPI == EPI_BIAS) {
                    v += p0[m];
                } else if (EPI == EPI_BNSILU) {
                    float z = v * p0[m] + p1[m]; v = z * sigm_(z);
                } else if (EPI == EPI_BNSILU_SIG || EPI == EPI_BNSILU_SIG_MUL) {
                    float z = v * p0[m] + p1[m]; z = z * sigm_(z); v = sigm_(z);
                } else if (EPI == EPI_GELU) {
                    float z = v + p0[m]; v = gelu_(z);
                }
                size_t oidx = (((size_t)bz * M + m) << 12) + n;
                if (OUTBF16) {
                    ((__bf16*)Outv)[oidx] = (__bf16)v;
                } else {
                    float* Of = (float*)Outv;
                    if (EPI == EPI_BNSILU_SIG_MUL) v *= Of[oidx];   // fuse f1*f2 in place
                    Of[oidx] = v;
                }
            }
        }
    }
}

// ======================= elementwise / small kernels =======================
__global__ void absdiff_kernel(const float* a, const float* b, float* o) {
    size_t i = (size_t)blockIdx.x * 256 + threadIdx.x;
    o[i] = fabsf(a[i] - b[i]);
}

__global__ void pool9h_kernel(const float* in, float* out) {
    size_t i = (size_t)blockIdx.x * 256 + threadIdx.x;
    int w = (int)(i & 63);
    size_t base = i - w;
    int lo = w - 4 < 0 ? 0 : w - 4;
    int hi = w + 4 > 63 ? 63 : w + 4;
    float s = 0.f;
    for (int j = lo; j <= hi; ++j) s += in[base + j];
    out[i] = s;
}

__global__ void pool9v_kernel(const float* in, float* out) {
    size_t i = (size_t)blockIdx.x * 256 + threadIdx.x;
    int h = (int)((i >> 6) & 63);
    size_t base = i - ((size_t)h << 6);
    int lo = h - 4 < 0 ? 0 : h - 4;
    int hi = h + 4 > 63 ? 63 : h + 4;
    float s = 0.f;
    for (int j = lo; j <= hi; ++j) s += in[base + ((size_t)j << 6)];
    out[i] = s * (1.0f / 81.0f);
}

__global__ void dwh_kernel(const float* in, const float* wt, const float* bias, float* out) {
    size_t i = (size_t)blockIdx.x * 256 + threadIdx.x;
    if (i >= (size_t)SH_ELEMS) return;
    int w = (int)(i % 58);
    size_t r = i / 58;                      // bc*64 + h
    int c = (int)((r >> 6) & 255);
    const float* src = in + (r << 6) + w;
    float s = bias[c];
    #pragma unroll
    for (int j = 0; j < 7; ++j) s += src[j] * wt[c * 7 + j];
    out[i] = s;
}

__global__ void dwv_kernel(const float* in, const float* wt, const float* bias, float* out) {
    size_t i = (size_t)blockIdx.x * 256 + threadIdx.x;
    if (i >= (size_t)SH_ELEMS) return;
    int w = (int)(i & 63);
    size_t r = i >> 6;                      // bc*58 + h
    int h = (int)(r % 58);
    size_t bc = r / 58;
    int c = (int)(bc & 255);
    const float* src = in + (bc << 12) + ((size_t)h << 6) + w;
    float s = bias[c];
    #pragma unroll
    for (int j = 0; j < 7; ++j) s += src[(size_t)j << 6] * wt[c * 7 + j];
    out[i] = s;
}

// batched (64x58)@(58x64) per (b,c)
__global__ __launch_bounds__(256)
void bmm_kernel(const float* __restrict__ Am, const float* __restrict__ Bm, float* __restrict__ Om) {
    __shared__ float Axs[64][60];
    __shared__ float Bxs[58][64];
    int t = threadIdx.x;
    size_t bc = blockIdx.x;
    const float* Ab = Am + bc * (64 * 58);
    const float* Bb = Bm + bc * (58 * 64);
    for (int i = t; i < 64 * 58; i += 256) Axs[i / 58][i % 58] = Ab[i];
    for (int i = t; i < 58 * 64; i += 256) Bxs[i >> 6][i & 63] = Bb[i];
    __syncthreads();
    int w = t & 63, h0 = (t >> 6) << 4;
    float acc[16];
    #pragma unroll
    for (int i = 0; i < 16; ++i) acc[i] = 0.f;
    for (int j = 0; j < 58; ++j) {
        float bv = Bxs[j][w];
        #pragma unroll
        for (int i = 0; i < 16; ++i) acc[i] += Axs[h0 + i][j] * bv;
    }
    float* Ob = Om + (bc << 12);
    #pragma unroll
    for (int i = 0; i < 16; ++i) Ob[((size_t)(h0 + i) << 6) + w] = acc[i];
}

// ---------------- LVC ----------------
__global__ __launch_bounds__(256)
void lvc_assign_kernel(const float* __restrict__ xproj, const float* __restrict__ cw,
                       const float* __restrict__ sc, float* __restrict__ Abuf,
                       float* __restrict__ AsumP) {
    __shared__ float Xs[64][257];
    __shared__ float Ss[64][68];
    __shared__ float x2s[64];
    int t = threadIdx.x;
    int b = blockIdx.x >> 6;
    int tile = blockIdx.x & 63;
    int n0 = tile << 6;
    for (int i = 0; i < 64; ++i) {
        int lin = t + (i << 8);
        int c = lin >> 6, p = lin & 63;
        Xs[p][c] = xproj[(((size_t)b * C_CH + c) << 12) + n0 + p];
    }
    __syncthreads();
    int p = t & 63, kq = t >> 6;
    if (kq == 0) {
        float s = 0.f;
        for (int c = 0; c < C_CH; ++c) { float v = Xs[p][c]; s += v * v; }
        x2s[p] = s;
    }
    float xc[16], c2[16];
    #pragma unroll
    for (int kk = 0; kk < 16; ++kk) { xc[kk] = 0.f; c2[kk] = 0.f; }
    for (int c = 0; c < C_CH; ++c) {
        float xv = Xs[p][c];
        #pragma unroll
        for (int kk = 0; kk < 16; ++kk) {
            float w = cw[(kq * 16 + kk) * C_CH + c];
            xc[kk] += xv * w;
            c2[kk] += w * w;
        }
    }
    __syncthreads();
    #pragma unroll
    for (int kk = 0; kk < 16; ++kk) {
        int k = kq * 16 + kk;
        Ss[p][k] = sc[k] * (x2s[p] + c2[kk] - 2.0f * xc[kk]);
    }
    __syncthreads();
    if (t < 64) {  // softmax over K for pixel p=t
        float mx = -1e30f;
        for (int k = 0; k < 64; ++k) mx = fmaxf(mx, Ss[t][k]);
        float sum = 0.f;
        for (int k = 0; k < 64; ++k) { float e = expf(Ss[t][k] - mx); Ss[t][k] = e; sum += e; }
        float inv = 1.0f / sum;
        float* dst = &Abuf[(((size_t)b << 12) + n0 + t) * 64];
        for (int k = 0; k < 64; ++k) { float a = Ss[t][k] * inv; Ss[t][k] = a; dst[k] = a; }
    }
    __syncthreads();
    if (t < 64) {  // per-tile partial column sums of A
        float s = 0.f;
        for (int pp = 0; pp < 64; ++pp) s += Ss[pp][t];
        AsumP[(((size_t)b << 6) + tile) * 64 + t] = s;
    }
}

__global__ void asum_reduce_kernel(const float* AsumP, float* Asum) {
    int t = blockIdx.x * 256 + threadIdx.x;  // b*64 + k, 1024 total
    int b = t >> 6, k = t & 63;
    float s = 0.f;
    for (int tile = 0; tile < 64; ++tile) s += AsumP[(((size_t)b << 6) + tile) * 64 + k];
    Asum[t] = s;
}

__global__ __launch_bounds__(256)
void lvc_agg_kernel(const float* __restrict__ xproj, const float* __restrict__ Abuf,
                    float* __restrict__ partial) {
    __shared__ float Xs[C_CH][66];
    __shared__ float As_[64][64];
    int t = threadIdx.x;                 // channel c
    int b = blockIdx.x >> 4, s = blockIdx.x & 15;
    float acc[64];
    #pragma unroll
    for (int k = 0; k < 64; ++k) acc[k] = 0.f;
    for (int ch = 0; ch < 4; ++ch) {
        int n0 = (s << 8) + (ch << 6);
        __syncthreads();
        for (int i = 0; i < 64; ++i) {
            int lin = t + (i << 8);
            int cc = lin >> 6, nl = lin & 63;
            Xs[cc][nl] = xproj[(((size_t)b * C_CH + cc) << 12) + n0 + nl];
        }
        for (int i = 0; i < 16; ++i) {
            int lin = t + (i << 8);
            int nl = lin >> 6, k = lin & 63;
            As_[nl][k] = Abuf[(((size_t)b << 12) + n0 + nl) * 64 + k];
        }
        __syncthreads();
        for (int nl = 0; nl < 64; ++nl) {
            float xv = Xs[t][nl];
            #pragma unroll
            for (int k = 0; k < 64; ++k) acc[k] += As_[nl][k] * xv;
        }
    }
    float* dst = &partial[((((size_t)b << 4) + s) * 64) * 256];
    for (int k = 0; k < 64; ++k) dst[(size_t)k * 256 + t] = acc[k];
}

__global__ void lvc_e_kernel(const float* partial, const float* Asum, const float* cw,
                             const float* bg, const float* bb, float* ebuf) {
    int t = threadIdx.x;  // c
    int b = blockIdx.x;
    float e = 0.f;
    for (int k = 0; k < 64; ++k) {
        float a = 0.f;
        for (int s = 0; s < 16; ++s)
            a += partial[((((size_t)b << 4) + s) * 64 + k) * 256 + t];
        a -= Asum[(b << 6) + k] * cw[k * 256 + t];
        float v = a * bg[k] + bb[k];
        e += fmaxf(v, 0.0f);
    }
    ebuf[b * 256 + t] = e * (1.0f / 64.0f);
}

__global__ void lvc_gam_kernel(const float* ebuf, const float* fcw, const float* fcb, float* gam) {
    __shared__ float es[256];
    int t = threadIdx.x, b = blockIdx.x;
    es[t] = ebuf[b * 256 + t];
    __syncthreads();
    float s = fcb[t];
    for (int c = 0; c < 256; ++c) s += es[c] * fcw[t * 256 + c];
    gam[b * 256 + t] = sigm_(s);
}

__global__ void combine_kernel(float* x, const float* f12, const float* xproj, const float* gam) {
    size_t i = (size_t)blockIdx.x * 256 + threadIdx.x;
    int bc = (int)(i >> 12);
    float xt = xproj[i] * (1.0f + gam[bc]);
    xt = fmaxf(xt, 0.0f);
    x[i] = x[i] * f12[i] + xt;
}

// ---------------- channels-first LayerNorm ----------------
__global__ __launch_bounds__(256)
void ln_cf_kernel(const float* __restrict__ in, float* __restrict__ out,
                  const float* __restrict__ g, const float* __restrict__ bta) {
    __shared__ float Xs[C_CH][65];
    __shared__ float red1[256], red2[256];
    __shared__ float mean_s[64], rstd_s[64];
    int t = threadIdx.x;
    int b = blockIdx.x >> 6;
    int n0 = (blockIdx.x & 63) << 6;
    for (int i = 0; i < 64; ++i) {
        int lin = t + (i << 8);
        int c = lin >> 6, p = lin & 63;
        Xs[c][p] = in[(((size_t)b * C_CH + c) << 12) + n0 + p];
    }
    __syncthreads();
    int p = t & 63, q = t >> 6;
    float s1 = 0.f, s2 = 0.f;
    for (int c = q * 64; c < q * 64 + 64; ++c) { float v = Xs[c][p]; s1 += v; s2 += v * v; }
    red1[(p << 2) + q] = s1; red2[(p << 2) + q] = s2;
    __syncthreads();
    if (q == 0) {
        float a1 = red1[p << 2] + red1[(p << 2) + 1] + red1[(p << 2) + 2] + red1[(p << 2) + 3];
        float a2 = red2[p << 2] + red2[(p << 2) + 1] + red2[(p << 2) + 2] + red2[(p << 2) + 3];
        float mu  = a1 * (1.0f / 256.0f);
        float var = a2 * (1.0f / 256.0f) - mu * mu;
        mean_s[p] = mu;
        rstd_s[p] = rsqrtf(var + 1e-6f);
    }
    __syncthreads();
    for (int i = 0; i < 64; ++i) {
        int lin = t + (i << 8);
        int c = lin >> 6, pp = lin & 63;
        float v = (Xs[c][pp] - mean_s[pp]) * rstd_s[pp] * g[c] + bta[c];
        out[(((size_t)b * C_CH + c) << 12) + n0 + pp] = v;
    }
}

__global__ void dw7x7_kernel(const float* in, const float* wt, const float* bias, float* out) {
    size_t i = (size_t)blockIdx.x * 256 + threadIdx.x;
    int w = (int)(i & 63), h = (int)((i >> 6) & 63);
    int c = (int)((i >> 12) & 255);
    size_t base = i - ((size_t)h << 6) - w;
    float s = bias[c];
    const float* wc = wt + c * 49;
    #pragma unroll
    for (int ki = 0; ki < 7; ++ki) {
        int hh = h - 3 + ki;
        if ((unsigned)hh > 63u) continue;
        #pragma unroll
        for (int kj = 0; kj < 7; ++kj) {
            int ww = w - 3 + kj;
            if ((unsigned)ww > 63u) continue;
            s += in[base + ((size_t)hh << 6) + ww] * wc[ki * 7 + kj];
        }
    }
    out[i] = s;
}

__global__ __launch_bounds__(256)
void gap_kernel(const float* in, float* out) {
    __shared__ float red[256];
    int t = threadIdx.x;
    size_t bc = blockIdx.x;
    const float* p = in + (bc << 12);
    float s = 0.f;
    for (int i = t; i < 4096; i += 256) s += p[i];
    red[t] = s;
    __syncthreads();
    for (int off = 128; off > 0; off >>= 1) { if (t < off) red[t] += red[t + off]; __syncthreads(); }
    if (t == 0) out[bc] = red[0] * (1.0f / 4096.0f);
}

__global__ void ca_mlp_kernel(const float* gap, const float* w1, const float* w2, float* cav) {
    __shared__ float gs[256];
    __shared__ float hs[8];
    int t = threadIdx.x, b = blockIdx.x;
    gs[t] = gap[b * 256 + t];
    __syncthreads();
    if (t < 8) {
        float s = 0.f;
        for (int c = 0; c < 256; ++c) s += w1[t * 256 + c] * gs[c];
        hs[t] = gelu_(s);
    }
    __syncthreads();
    float s = 0.f;
    #pragma unroll
    for (int j = 0; j < 8; ++j) s += w2[t * 8 + j] * hs[j];
    cav[b * 256 + t] = sigm_(s);
}

__global__ void final_kernel(const float* y, const float* cav, const float* res, float* out) {
    size_t i = (size_t)blockIdx.x * 256 + threadIdx.x;
    int bc = (int)(i >> 12);
    out[i] = y[i] * cav[bc] + res[i];
}

// ======================= host side =======================
extern "C" void kernel_launch(void* const* d_in, const int* in_sizes, int n_in,
                              void* d_out, int out_size, void* d_ws, size_t ws_size,
                              hipStream_t stream)
{
    const float* x1 = (const float*)d_in[0];
    const float* x2 = (const float*)d_in[1];
    const float* conv1_w = (const float*)d_in[2];
    const float* conv1_g = (const float*)d_in[3];
    const float* conv1_b = (const float*)d_in[4];
    const float* h_w = (const float*)d_in[5];
    const float* h_b = (const float*)d_in[6];
    const float* v_w = (const float*)d_in[7];
    const float* v_b = (const float*)d_in[8];
    const float* conv0_w = (const float*)d_in[9];
    const float* conv0_b = (const float*)d_in[10];
    const float* q_w = (const float*)d_in[11];
    const float* q_g = (const float*)d_in[12];
    const float* q_b = (const float*)d_in[13];
    const float* k_w = (const float*)d_in[14];
    const float* k_g = (const float*)d_in[15];
    const float* k_b = (const float*)d_in[16];
    const float* proj_w = (const float*)d_in[17];
    const float* proj_b = (const float*)d_in[18];
    const float* cw = (const float*)d_in[19];
    const float* sc = (const float*)d_in[20];
    const float* bn_g = (const float*)d_in[21];
    const float* bn_b = (const float*)d_in[22];
    const float* fc_w = (const float*)d_in[23];
    const float* fc_b = (const float*)d_in[24];
    const float* ln2_g = (const float*)d_in[25];
    const float* ln2_b = (const float*)d_in[26];
    const float* dw_w = (const float*)d_in[27];
    const float* dw_b = (const float*)d_in[28];
    const float* dln_g = (const float*)d_in[29];
    const float* dln_b = (const float*)d_in[30];
    const float* pw1_w = (const float*)d_in[31];
    const float* pw1_b = (const float*)d_in[32];
    const float* pw2_w = (const float*)d_in[33];
    const float* pw2_b = (const float*)d_in[34];
    const float* ca_w1 = (const float*)d_in[35];
    const float* ca_w2 = (const float*)d_in[36];

    const size_t SB  = (size_t)S_ELEMS  * 4;   // 67108864
    const size_t SHB = (size_t)SH_ELEMS * 4;   // 60817408
    const size_t AB  = (size_t)BATCH * NPIX * 64 * 4;  // 16777216

    char* ws = (char*)d_ws;
    float* bufX = (float*)(ws);
    float* bufP = (float*)(ws + SB);
    float* bufQ = (float*)(ws + 2 * SB);
    char*  regR = ws + 3 * SB;                 // shared region: edges / A / hidden
    float* bufXh = (float*)(regR);
    float* bufXw = (float*)(regR + SHB);
    float* bufYh = (float*)(regR + 2 * SHB);
    float* bufYw = (float*)(regR + 3 * SHB);
    float* Abuf    = (float*)(regR);           // alive after edges die
    float* partial = (float*)(regR + AB);
    void*  hidden  = (void*)(regR);            // bf16, alive after LVC dies
    char* small = ws + 3 * SB + 4 * SHB;
    float* Asum  = (float*)(small);
    float* ebuf  = (float*)(small + 4096);
    float* gam   = (float*)(small + 4096 + 16384);
    float* gapb  = (float*)(small + 4096 + 2 * 16384);
    float* cav   = (float*)(small + 4096 + 3 * 16384);
    float* AsumP = (float*)(small + 4096 + 4 * 16384);

    dim3 blk(256);
    const int gS  = S_ELEMS / 256;
    const int gSH = (SH_ELEMS + 255) / 256;
    dim3 ggC(NPIX / BN, 256 / BM, BATCH);      // M=256 GEMMs
    dim3 ggH(NPIX / BN, 1024 / BM, BATCH);     // pw1

    // 1) x = |x1-x2|
    absdiff_kernel<<<gS, blk, 0, stream>>>(x1, x2, bufX);
    // 2) avgpool9(x) -> bufQ
    pool9h_kernel<<<gS, blk, 0, stream>>>(bufX, bufP);
    pool9v_kernel<<<gS, blk, 0, stream>>>(bufP, bufQ);
    // 3) x1p = silu(bn(conv1(ap_x))) -> bufP
    gemm1x1_kernel<EPI_BNSILU, false, false, false><<<ggC, blk, 0, stream>>>(
        conv1_w, bufQ, nullptr, bufP, conv1_g, conv1_b, 256, 256);
    // 4) xh, xw
    dwh_kernel<<<gSH, blk, 0, stream>>>(bufP, h_w, h_b, bufXh);
    dwv_kernel<<<gSH, blk, 0, stream>>>(bufP, v_w, v_b, bufXw);
    // 5) y1 = conv0(concat(x1,x2)) + b -> bufP
    gemm1x1_kernel<EPI_BIAS, true, false, false><<<ggC, blk, 0, stream>>>(
        conv0_w, x1, x2, bufP, conv0_b, nullptr, 256, 512);
    // 6) avgpool9(y1) -> bufP
    pool9h_kernel<<<gS, blk, 0, stream>>>(bufP, bufQ);
    pool9v_kernel<<<gS, blk, 0, stream>>>(bufQ, bufP);
    // 7) y1p -> bufQ
    gemm1x1_kernel<EPI_BNSILU, false, false, false><<<ggC, blk, 0, stream>>>(
        conv1_w, bufP, nullptr, bufQ, conv1_g, conv1_b, 256, 256);
    // 8) yh, yw
    dwh_kernel<<<gSH, blk, 0, stream>>>(bufQ, h_w, h_b, bufYh);
    dwv_kernel<<<gSH, blk, 0, stream>>>(bufQ, v_w, v_b, bufYw);
    // 9) mm1 = xh @ yw -> bufP ; 10) f1 = sigmoid(silu(bn(q(mm1)))) -> bufQ
    bmm_kernel<<<BATCH * C_CH, blk, 0, stream>>>(bufXh, bufYw, bufP);
    gemm1x1_kernel<EPI_BNSILU_SIG, false, false, false><<<ggC, blk, 0, stream>>>(
        q_w, bufP, nullptr, bufQ, q_g, q_b, 256, 256);
    // 11) mm2 = yh @ xw -> bufP ; 12) bufQ = f1 * f2 (fused)
    bmm_kernel<<<BATCH * C_CH, blk, 0, stream>>>(bufYh, bufXw, bufP);
    gemm1x1_kernel<EPI_BNSILU_SIG_MUL, false, false, false><<<ggC, blk, 0, stream>>>(
        k_w, bufP, nullptr, bufQ, k_g, k_b, 256, 256);
    // 13) xproj = gelu(proj(concat)+b) -> bufP
    gemm1x1_kernel<EPI_GELU, true, false, false><<<ggC, blk, 0, stream>>>(
        proj_w, x1, x2, bufP, proj_b, nullptr, 256, 512);
    // 14) assignment softmax + partial column sums
    lvc_assign_kernel<<<BATCH * 64, blk, 0, stream>>>(bufP, cw, sc, Abuf, AsumP);
    asum_reduce_kernel<<<4, blk, 0, stream>>>(AsumP, Asum);
    // 15) agg partials ; 16) e ; 17) gamma
    lvc_agg_kernel<<<BATCH * 16, blk, 0, stream>>>(bufP, Abuf, partial);
    lvc_e_kernel<<<BATCH, blk, 0, stream>>>(partial, Asum, cw, bn_g, bn_b, ebuf);
    lvc_gam_kernel<<<BATCH, blk, 0, stream>>>(ebuf, fc_w, fc_b, gam);
    // 18) out0 = x*f1*f2 + relu(xproj*(1+gam))  (in-place in bufX)
    combine_kernel<<<gS, blk, 0, stream>>>(bufX, bufQ, bufP, gam);
    // 19) out1 = LN(out0) -> bufP
    ln_cf_kernel<<<BATCH * 64, blk, 0, stream>>>(bufX, bufP, ln2_g, ln2_b);
    // 20-21) dcab: dw7x7 -> bufX ; LN -> bufQ
    dw7x7_kernel<<<gS, blk, 0, stream>>>(bufP, dw_w, dw_b, bufX);
    ln_cf_kernel<<<BATCH * 64, blk, 0, stream>>>(bufX, bufQ, dln_g, dln_b);
    // 22) hidden = gelu(pw1 + b1) (bf16) ; 23) t3 = pw2(hidden)+b2 -> bufX
    gemm1x1_kernel<EPI_GELU, false, false, true><<<ggH, blk, 0, stream>>>(
        pw1_w, bufQ, nullptr, hidden, pw1_b, nullptr, 1024, 256);
    gemm1x1_kernel<EPI_BIAS, false, true, false><<<ggC, blk, 0, stream>>>(
        pw2_w, hidden, nullptr, bufX, pw2_b, nullptr, 256, 1024);
    // 24-26) channel attention + residual
    gap_kernel<<<BATCH * C_CH, blk, 0, stream>>>(bufX, gapb);
    ca_mlp_kernel<<<BATCH, blk, 0, stream>>>(gapb, ca_w1, ca_w2, cav);
    final_kernel<<<gS, blk, 0, stream>>>(bufX, cav, bufP, (float*)d_out);
}
